// simpleKT_SR_80358838108825
// MI455X (gfx1250) — matrix operand order repaired and checked by reading, not compiled, not run on hardware
//
#include <hip/hip_runtime.h>
#include <hip/hip_bf16.h>
#include <math.h>

// ---------------- types ----------------
typedef __bf16 bf16_t;
typedef float        v8f   __attribute__((ext_vector_type(8)));
typedef bf16_t       v16bf __attribute__((ext_vector_type(16)));
typedef bf16_t       v8bf  __attribute__((ext_vector_type(8)));
typedef unsigned int v4u   __attribute__((ext_vector_type(4)));
typedef int          v8i   __attribute__((ext_vector_type(8)));
typedef int          v4i   __attribute__((ext_vector_type(4)));

#define BB 8
#define SS 1024
#define DD 1024
#define HH 16
#define DK 64
#define DF 4096

static __device__ __forceinline__ v16bf cat8(v8bf a, v8bf b) {
  return __builtin_shufflevector(a, b, 0,1,2,3,4,5,6,7,8,9,10,11,12,13,14,15);
}
static __device__ __forceinline__ v8f wmma_bf16(v16bf a, v16bf b, v8f c) {
  // D = A(16x32 bf16) x B(32x16 bf16) + C(16x16 f32)
  return __builtin_amdgcn_wmma_f32_16x16x32_bf16(false, a, false, b, (short)0, c, false, false);
}

// ---------------- TDM: DMA a 2-D bf16 tile global -> LDS ----------------
// D# layout per CDNA5 ISA ch.8 (CDNA5 ISA). 2-D tensor: groups 2/3 zero.
// This toolchain exposes the 6-arg builtin: (v4u g0, v8i g1, v4i g2, v4i g3, v8i extra, i32 cpol).
static __device__ __forceinline__ void tdm_load_2d(unsigned lds_addr, const bf16_t* gptr,
                                                   unsigned tensor_d0, unsigned tensor_d1,
                                                   unsigned tile_d0, unsigned tile_d1,
                                                   unsigned stride_elems, unsigned padbits) {
  unsigned long long ga = (unsigned long long)(size_t)gptr;
  v4u g0;
  g0[0] = 1u;                                              // count=1, user mode
  g0[1] = lds_addr;                                        // lds_addr [63:32]
  g0[2] = (unsigned)(ga & 0xffffffffull);                  // global_addr lo
  g0[3] = (unsigned)((ga >> 32) & 0x01fffffful) | (2u << 30);  // global_addr hi | type=2
  v8i g1;
  g1[0] = (int)((1u << 16) | padbits);                     // data_size=2B (+pad cfg)
  g1[1] = (int)((tensor_d0 & 0xffffu) << 16);              // [63:48] = tensor_dim0 lo
  g1[2] = (int)((tensor_d0 >> 16) | ((tensor_d1 & 0xffffu) << 16)); // td0 hi | td1 lo
  g1[3] = (int)((tensor_d1 >> 16) | (tile_d0 << 16));      // td1 hi | tile_dim0
  g1[4] = (int)(tile_d1 & 0xffffu);                        // tile_dim1 | tile_dim2=0
  g1[5] = (int)stride_elems;                               // tensor_dim0_stride lo
  g1[6] = 0;                                               // stride hi | dim1_stride lo
  g1[7] = 0;                                               // dim1_stride hi
  v4i z4 = {0, 0, 0, 0};
  v8i z8 = {0, 0, 0, 0, 0, 0, 0, 0};
  __builtin_amdgcn_tensor_load_to_lds(g0, g1, z4, z4, z8, 0);
}
// pad: insert 4 DWORDs (16B) after every 16 DWORDs (64B row) -> LDS row stride 80B (40 bf16)
#define TDM_PAD_64B_TO_80B ((1u << 20) | (3u << 22) | (3u << 25))

// ---------------- f32 -> bf16 cast (weights) ----------------
__global__ __launch_bounds__(256) void cast_kernel(const float* __restrict__ in,
                                                   bf16_t* __restrict__ out, int n) {
  int i = blockIdx.x * blockDim.x + threadIdx.x;
  const int stride = gridDim.x * blockDim.x;
  for (; i < n; i += stride) out[i] = (bf16_t)in[i];
}

// ---------------- positional encoding + add ----------------
__global__ __launch_bounds__(256) void pe_add_kernel(const float* __restrict__ qe,
                                                     const float* __restrict__ qa,
                                                     float* __restrict__ x,
                                                     bf16_t* __restrict__ xb,
                                                     bf16_t* __restrict__ yb,
                                                     int total) {
  int i = blockIdx.x * blockDim.x + threadIdx.x;
  const int stride = gridDim.x * blockDim.x;
  for (; i < total; i += stride) {
    int d = i & (DD - 1);
    int s = (i >> 10) & (SS - 1);
    int j = d >> 1;
    float ang = (float)s * __expf((float)(2 * j) * (-9.2103403719761836f / (float)DD));
    float pe = (d & 1) ? __cosf(ang) : __sinf(ang);
    float xv = qe[i] + pe;
    x[i] = xv;
    xb[i] = (bf16_t)xv;
    yb[i] = (bf16_t)(qa[i] + pe);
  }
}

// ---------------- WMMA bf16 GEMM with TDM double-buffered LDS staging ----------------
// C[M,N] = A[M,K] * W[N,K]^T + bias ; A,W bf16 in global. block=256 (8 waves), tile 128x128, K-step 32.
__global__ __launch_bounds__(256) void gemm_kernel(const bf16_t* __restrict__ A,
                                                   const bf16_t* __restrict__ W,
                                                   const float* __restrict__ bias,
                                                   float* __restrict__ outF,
                                                   bf16_t* __restrict__ outB,
                                                   int M, int N, int K, int relu) {
  __shared__ __align__(16) bf16_t As[2][128][40];   // TDM-padded: 64B row + 16B pad
  __shared__ __align__(16) bf16_t Ws[2][128][40];
  const int tid = threadIdx.x;
  const int w  = tid >> 5;
  const int l  = tid & 31;
  const int lh = l >> 4;
  const int lm = l & 15;
  const int m0 = blockIdx.x * 128;
  const int n0 = blockIdx.y * 128;

  v8f acc[8];
#pragma unroll
  for (int i = 0; i < 8; ++i)
#pragma unroll
    for (int r = 0; r < 8; ++r) acc[i][r] = 0.0f;

  const bf16_t* abase = A + (size_t)m0 * K;
  const bf16_t* wbase = W + (size_t)n0 * K;
  const int nk = K >> 5;

  if (w == 0) {   // wave 0 drives the tensor DMA engine (EXEC ignored by TDM)
    tdm_load_2d((unsigned)(size_t)&As[0][0][0], abase, (unsigned)K, 128u, 32u, 128u,
                (unsigned)K, TDM_PAD_64B_TO_80B);
    tdm_load_2d((unsigned)(size_t)&Ws[0][0][0], wbase, (unsigned)K, 128u, 32u, 128u,
                (unsigned)K, TDM_PAD_64B_TO_80B);
  }

  int cur = 0;
  for (int kt = 0; kt < nk; ++kt) {
    if (w == 0) __builtin_amdgcn_s_wait_tensorcnt((short)0);
    __syncthreads();                                     // publish buffer `cur` to all waves
    if (w == 0 && (kt + 1) < nk) {                       // DMA next K-tile while computing
      int koff = (kt + 1) * 32;
      tdm_load_2d((unsigned)(size_t)&As[cur ^ 1][0][0], abase + koff, (unsigned)K, 128u,
                  32u, 128u, (unsigned)K, TDM_PAD_64B_TO_80B);
      tdm_load_2d((unsigned)(size_t)&Ws[cur ^ 1][0][0], wbase + koff, (unsigned)K, 128u,
                  32u, 128u, (unsigned)K, TDM_PAD_64B_TO_80B);
    }

    // A fragment: row = 16w + lm ; element e -> k = e + 16*lh
    const bf16_t* ap = &As[cur][w * 16 + lm][0];
    v16bf afrag = cat8(*(const v8bf*)(ap + 16 * lh), *(const v8bf*)(ap + 16 * lh + 8));
#pragma unroll
    for (int nt = 0; nt < 8; ++nt) {
      // B fragment: n = 16nt + lm ; element e -> k = e + 16*lh
      const bf16_t* bp = &Ws[cur][nt * 16 + lm][0];
      v16bf bfrag = cat8(*(const v8bf*)(bp + 16 * lh), *(const v8bf*)(bp + 16 * lh + 8));
      acc[nt] = wmma_bf16(afrag, bfrag, acc[nt]);
    }
    cur ^= 1;
  }

  const int mrow = m0 + w * 16 + 8 * lh;   // C layout: row = r + 8*lh, col = lm
#pragma unroll
  for (int nt = 0; nt < 8; ++nt) {
    int col = n0 + nt * 16 + lm;
    float bv = bias ? bias[col] : 0.0f;
#pragma unroll
    for (int r = 0; r < 8; ++r) {
      float vv = acc[nt][r] + bv;
      if (relu) vv = fmaxf(vv, 0.0f);
      size_t off = (size_t)(mrow + r) * N + col;
      if (outF) outF[off] = vv;
      if (outB) outB[off] = (bf16_t)vv;
    }
  }
}

// ---------------- flash attention (strict causal, q == k, row 0 zeroed), bf16 in/out ----------------
__global__ __launch_bounds__(256) void attn_kernel(const bf16_t* __restrict__ qk,
                                                   const bf16_t* __restrict__ vv,
                                                   bf16_t* __restrict__ out) {
  __shared__ __align__(16) bf16_t Ks[32][72];      // [key j'][dk]
  __shared__ __align__(16) bf16_t Vs[64][40];      // transposed: [dk][key j']
  __shared__ __align__(16) bf16_t Ps[8][16][40];   // per-wave P tile [row][j']
  const int tid = threadIdx.x;
  const int w  = tid >> 5;
  const int l  = tid & 31;
  const int lh = l >> 4;
  const int lm = l & 15;
  const int bh = blockIdx.y;
  const int b  = bh >> 4;
  const int h  = bh & (HH - 1);
  const int qb = blockIdx.x * 128;
  const int qr0 = qb + w * 16;

  const bf16_t* qp = qk + (size_t)b * SS * DD + (size_t)h * DK;
  const bf16_t* vp = vv + (size_t)b * SS * DD + (size_t)h * DK;

  // Q A-fragments (K-dim = 64 -> two 16x32 chunks) loaded directly in A layout
  const bf16_t* qrow = qp + (size_t)(qr0 + lm) * DD;
  v16bf Aq[2];
#pragma unroll
  for (int c = 0; c < 2; ++c)
#pragma unroll
    for (int e = 0; e < 16; ++e) {
      int dk = 32 * c + e + 16 * lh;
      Aq[c][e] = qrow[dk];
    }

  v8f o[4];
  float mrow[8], ssum[8];
#pragma unroll
  for (int t4 = 0; t4 < 4; ++t4)
#pragma unroll
    for (int r = 0; r < 8; ++r) o[t4][r] = 0.0f;
#pragma unroll
  for (int r = 0; r < 8; ++r) { mrow[r] = -3.0e38f; ssum[r] = 0.0f; }

  const int jmax = qb + 126;
  for (int j0 = 0; j0 <= jmax; j0 += 32) {
    __syncthreads();
    {   // cooperative load of 32 keys x 64 dk (K row-major, V transposed)
      int jj  = tid >> 3;
      int dk0 = (tid & 7) * 8;
      const bf16_t* kr = qp + (size_t)(j0 + jj) * DD + dk0;
      const bf16_t* vr = vp + (size_t)(j0 + jj) * DD + dk0;
#pragma unroll
      for (int e = 0; e < 8; ++e) {
        Ks[jj][dk0 + e] = kr[e];
        Vs[dk0 + e][jj] = vr[e];
      }
    }
    __syncthreads();

    if (j0 <= qr0 + 14) {   // wave-uniform causal bound
      v8f s0, s1;
#pragma unroll
      for (int r = 0; r < 8; ++r) { s0[r] = 0.0f; s1[r] = 0.0f; }
      {  // keys j0 + 0..15
        const bf16_t* kp = &Ks[lm][0];
        v16bf b0 = cat8(*(const v8bf*)(kp + 16 * lh), *(const v8bf*)(kp + 16 * lh + 8));
        v16bf b1 = cat8(*(const v8bf*)(kp + 32 + 16 * lh), *(const v8bf*)(kp + 32 + 16 * lh + 8));
        s0 = wmma_bf16(Aq[0], b0, s0);
        s0 = wmma_bf16(Aq[1], b1, s0);
      }
      {  // keys j0 + 16..31
        const bf16_t* kp = &Ks[16 + lm][0];
        v16bf b0 = cat8(*(const v8bf*)(kp + 16 * lh), *(const v8bf*)(kp + 16 * lh + 8));
        v16bf b1 = cat8(*(const v8bf*)(kp + 32 + 16 * lh), *(const v8bf*)(kp + 32 + 16 * lh + 8));
        s1 = wmma_bf16(Aq[0], b0, s1);
        s1 = wmma_bf16(Aq[1], b1, s1);
      }
      // online softmax per row (row r+8*lh lives across the 16 lanes of a half-wave)
#pragma unroll
      for (int r = 0; r < 8; ++r) {
        int i = qr0 + r + 8 * lh;
        float v0 = s0[r] * 0.125f; if (j0 + lm >= i)      v0 = -1.0e32f;
        float v1 = s1[r] * 0.125f; if (j0 + 16 + lm >= i) v1 = -1.0e32f;
        float t = fmaxf(v0, v1);
#pragma unroll
        for (int off = 8; off > 0; off >>= 1) t = fmaxf(t, __shfl_xor(t, off, 32));
        float mnew = fmaxf(mrow[r], t);
        float corr = __expf(mrow[r] - mnew);
        mrow[r] = mnew;
        float p0 = __expf(v0 - mnew);
        float p1 = __expf(v1 - mnew);
        float rs = p0 + p1;
#pragma unroll
        for (int off = 8; off > 0; off >>= 1) rs += __shfl_xor(rs, off, 32);
        ssum[r] = ssum[r] * corr + rs;
        o[0][r] *= corr; o[1][r] *= corr; o[2][r] *= corr; o[3][r] *= corr;
        Ps[w][r + 8 * lh][lm]      = (bf16_t)p0;
        Ps[w][r + 8 * lh][16 + lm] = (bf16_t)p1;
      }
      // P (16x32) as A-fragment (same-wave LDS RAW: DS ops in-order per wave)
      const bf16_t* pp = &Ps[w][lm][0];
      v16bf Ap = cat8(*(const v8bf*)(pp + 16 * lh), *(const v8bf*)(pp + 16 * lh + 8));
#pragma unroll
      for (int t4 = 0; t4 < 4; ++t4) {
        const bf16_t* vb = &Vs[16 * t4 + lm][0];
        v16bf Bv = cat8(*(const v8bf*)(vb + 16 * lh), *(const v8bf*)(vb + 16 * lh + 8));
        o[t4] = wmma_bf16(Ap, Bv, o[t4]);
      }
    }
  }

  bf16_t* op = out + (size_t)b * SS * DD + (size_t)h * DK;
#pragma unroll
  for (int t4 = 0; t4 < 4; ++t4) {
    int dk = 16 * t4 + lm;
#pragma unroll
    for (int r = 0; r < 8; ++r) {
      int srow = qr0 + r + 8 * lh;
      float val = (srow == 0) ? 0.0f : (o[t4][r] / ssum[r]);   // zero_pad first query row
      op[(size_t)srow * DD + dk] = (bf16_t)val;
    }
  }
}

// ---------------- residual + LayerNorm: x = LN(x + t) * g + b ; also emits bf16 copy ----------------
__global__ __launch_bounds__(256) void ln_kernel(float* __restrict__ x,
                                                 bf16_t* __restrict__ xb,
                                                 const float* __restrict__ t,
                                                 const float* __restrict__ g,
                                                 const float* __restrict__ be) {
  __shared__ float red[16];
  const int row = blockIdx.x;
  float* xr = x + (size_t)row * DD;
  bf16_t* xbr = xb + (size_t)row * DD;
  const float* tr = t + (size_t)row * DD;
  float v[4];
  float s = 0.0f, sq = 0.0f;
#pragma unroll
  for (int i = 0; i < 4; ++i) {
    int c = threadIdx.x + 256 * i;
    v[i] = xr[c] + tr[c];
    s += v[i];
    sq += v[i] * v[i];
  }
#pragma unroll
  for (int off = 16; off > 0; off >>= 1) {
    s  += __shfl_xor(s, off, 32);
    sq += __shfl_xor(sq, off, 32);
  }
  int wv = threadIdx.x >> 5;
  if ((threadIdx.x & 31) == 0) { red[wv] = s; red[8 + wv] = sq; }
  __syncthreads();
  float S = 0.0f, SQ = 0.0f;
#pragma unroll
  for (int i = 0; i < 8; ++i) { S += red[i]; SQ += red[8 + i]; }
  float mean = S * (1.0f / (float)DD);
  float var  = SQ * (1.0f / (float)DD) - mean * mean;
  float rstd = rsqrtf(var + 1e-5f);
#pragma unroll
  for (int i = 0; i < 4; ++i) {
    int c = threadIdx.x + 256 * i;
    float o = (v[i] - mean) * rstd * g[c] + be[c];
    xr[c] = o;
    xbr[c] = (bf16_t)o;
  }
}

// ---------------- launch ----------------
extern "C" void kernel_launch(void* const* d_in, const int* in_sizes, int n_in,
                              void* d_out, int out_size, void* d_ws, size_t ws_size,
                              hipStream_t stream) {
  const float* qe   = (const float*)d_in[0];
  const float* qa   = (const float*)d_in[1];
  const float* Wk   = (const float*)d_in[2];
  const float* bk   = (const float*)d_in[3];
  const float* Wv   = (const float*)d_in[4];
  const float* bv   = (const float*)d_in[5];
  const float* Wo   = (const float*)d_in[6];
  const float* bo   = (const float*)d_in[7];
  const float* ln1g = (const float*)d_in[8];
  const float* ln1b = (const float*)d_in[9];
  const float* W1   = (const float*)d_in[10];
  const float* b1   = (const float*)d_in[11];
  const float* W2   = (const float*)d_in[12];
  const float* b2   = (const float*)d_in[13];
  const float* ln2g = (const float*)d_in[14];
  const float* ln2b = (const float*)d_in[15];

  const size_t NE  = (size_t)BB * SS * DD;   // 8M
  const size_t NF  = (size_t)BB * SS * DF;   // 32M
  const size_t WDD = (size_t)2 * DD * DD;    // per-weight (L=2) elems, D x D
  const size_t WDF = (size_t)2 * DF * DD;

  char* ws = (char*)d_ws;
  size_t off = 0;
  float*  x    = (float*)(ws + off);  off += NE * 4;
  float*  tbuf = (float*)(ws + off);  off += NE * 4;
  bf16_t* xb   = (bf16_t*)(ws + off); off += NE * 2;
  bf16_t* yb   = (bf16_t*)(ws + off); off += NE * 2;
  bf16_t* qkb  = (bf16_t*)(ws + off); off += NE * 2;
  bf16_t* vb   = (bf16_t*)(ws + off); off += NE * 2;
  bf16_t* ab   = (bf16_t*)(ws + off); off += NE * 2;
  bf16_t* hb   = (bf16_t*)(ws + off); off += NF * 2;
  bf16_t* Wkb  = (bf16_t*)(ws + off); off += WDD * 2;
  bf16_t* Wvb  = (bf16_t*)(ws + off); off += WDD * 2;
  bf16_t* Wob  = (bf16_t*)(ws + off); off += WDD * 2;
  bf16_t* W1b  = (bf16_t*)(ws + off); off += WDF * 2;
  bf16_t* W2b  = (bf16_t*)(ws + off); off += WDF * 2;

  const int M = BB * SS;   // 8192
  dim3 blk(256);

  // one-time per-launch weight casts (L2-resident afterwards)
  cast_kernel<<<2048, blk, 0, stream>>>(Wk, Wkb, (int)WDD);
  cast_kernel<<<2048, blk, 0, stream>>>(Wv, Wvb, (int)WDD);
  cast_kernel<<<2048, blk, 0, stream>>>(Wo, Wob, (int)WDD);
  cast_kernel<<<4096, blk, 0, stream>>>(W1, W1b, (int)WDF);
  cast_kernel<<<4096, blk, 0, stream>>>(W2, W2b, (int)WDF);

  pe_add_kernel<<<2048, blk, 0, stream>>>(qe, qa, x, xb, yb, (int)NE);

  for (int lyr = 0; lyr < 2; ++lyr) {
    const size_t wdd = (size_t)lyr * DD * DD;
    const size_t wdf = (size_t)lyr * DF * DD;
    // q = k = x @ Wk^T + bk   (bf16 out for attention)
    gemm_kernel<<<dim3(M / 128, DD / 128), blk, 0, stream>>>(
        xb, Wkb + wdd, bk + lyr * DD, nullptr, qkb, M, DD, DD, 0);
    // v = y @ Wv^T + bv
    gemm_kernel<<<dim3(M / 128, DD / 128), blk, 0, stream>>>(
        yb, Wvb + wdd, bv + lyr * DD, nullptr, vb, M, DD, DD, 0);
    // attention
    attn_kernel<<<dim3(SS / 128, BB * HH), blk, 0, stream>>>(qkb, vb, ab);
    // attn @ Wo^T + bo (f32 for residual)
    gemm_kernel<<<dim3(M / 128, DD / 128), blk, 0, stream>>>(
        ab, Wob + wdd, bo + lyr * DD, tbuf, nullptr, M, DD, DD, 0);
    // x = LN(x + attn_out) (+ bf16 copy)
    ln_kernel<<<M, blk, 0, stream>>>(x, xb, tbuf, ln1g + lyr * DD, ln1b + lyr * DD);
    // h = relu(x @ W1^T + b1) -> bf16
    gemm_kernel<<<dim3(M / 128, DF / 128), blk, 0, stream>>>(
        xb, W1b + wdf, b1 + lyr * DF, nullptr, hb, M, DF, DD, 1);
    // ff = h @ W2^T + b2 (f32 for residual)
    gemm_kernel<<<dim3(M / 128, DD / 128), blk, 0, stream>>>(
        hb, W2b + wdf, b2 + lyr * DD, tbuf, nullptr, M, DD, DF, 0);
    // x = LN(x + ff) (+ bf16 copy)
    ln_kernel<<<M, blk, 0, stream>>>(x, xb, tbuf, ln2g + lyr * DD, ln2b + lyr * DD);
  }

  (void)hipMemcpyAsync(d_out, x, NE * sizeof(float), hipMemcpyDeviceToDevice, stream);
}
